// GRUForecaster_1752346657204
// MI455X (gfx1250) — matrix operand order repaired and checked by reading, not compiled, not run on hardware
//
#include <hip/hip_runtime.h>
#include <hip/hip_bf16.h>

// ---------------------------------------------------------------------------
// 2-layer GRU (PyTorch semantics) + linear head for MI455X (gfx1250).
//  * GEMMs: v_wmma_f32_16x16x32_bf16 (bf16 operands, fp32 accumulation)
//  * Block = one 16-wide hidden-column strip; gate weight strips staged in
//    LDS via global_load_async_to_lds_b128 (ASYNCcnt), shared by all 8 waves.
//  * Wave = 32x16 output tile (2 M-tiles reuse each B fragment).
//  * Master h state fp32; bf16 shadow feeds the matrix units.
// ---------------------------------------------------------------------------

typedef __attribute__((ext_vector_type(16))) __bf16 v16bf;
typedef __attribute__((ext_vector_type(8)))  __bf16 v8bf;
typedef __attribute__((ext_vector_type(8)))  float  v8f;

#define GRU_H 512
#define GRU_B 256
#define GRU_T 168
#define GRU_I 7
#define GRU_OUT 24

// Use the CDNA5 async global->LDS DMA path (tracked by ASYNCcnt).
#define USE_ASYNC_LDS 1

// 16-byte global -> LDS copy.
// ISA 15.18.3 / CDNA5 ISA: GLOBAL_LOAD_ASYNC_TO_LDS_B128
//   LDS address comes from the VDST VGPR (plus offset), global address from VADDR
// VDST carries the per-lane LDS byte offset = low 32 bits of the generic ptr.
__device__ __forceinline__ void cp16_to_lds(const __bf16* g, __bf16* l) {
#if USE_ASYNC_LDS
  unsigned int loff = (unsigned int)(uintptr_t)l;        // LDS byte offset
  unsigned long long ga = (unsigned long long)(uintptr_t)g;
  asm volatile("global_load_async_to_lds_b128 %0, %1, off"
               :
               : "v"(loff), "v"(ga)
               : "memory");
#else
  *(v8bf*)l = *(const v8bf*)g;                           // ds_store path
#endif
}

__device__ __forceinline__ void stage_fence() {
#if USE_ASYNC_LDS
#if __has_builtin(__builtin_amdgcn_s_wait_asynccnt)
  __builtin_amdgcn_s_wait_asynccnt(0);
#else
  asm volatile("s_wait_asynccnt 0" ::: "memory");
#endif
#endif
  __syncthreads();
}

// ---- fragment loaders ------------------------------------------------------
// A matrix (16x32, MxK), 16-bit layout (ISA 7.12.2): lane holds row (lane&15),
// K chunks {lh*8..+7} and {lh*8+16..+23}, lh = lane>>4.
__device__ __forceinline__ v16bf load_a_frag(const __bf16* base, int ldk,
                                             int row, int kk, int lh) {
  const __bf16* p = base + (size_t)row * ldk + kk + lh * 8;
  union { v16bf v; v8bf h[2]; } u;
  u.h[0] = *(const v8bf*)(p);
  u.h[1] = *(const v8bf*)(p + 16);
  return u.v;
}

// B matrix (32x16, KxN): lane holds column (lane&15), K chunks {lh*8..+7}
// and {lh*8+16..+23}. Weight strips live in LDS as [strip][16][512] row-major K.
__device__ __forceinline__ v16bf load_b_lds(const __bf16* sW, int strip,
                                            int ln, int kk, int lh) {
  const __bf16* p = sW + ((size_t)(strip * 16 + ln)) * GRU_H + kk + lh * 8;
  union { v16bf v; v8bf h[2]; } u;
  u.h[0] = *(const v8bf*)(p);
  u.h[1] = *(const v8bf*)(p + 16);
  return u.v;
}

__device__ __forceinline__ v8f wmma_bf16(v16bf a, v16bf b, v8f c) {
  return __builtin_amdgcn_wmma_f32_16x16x32_bf16(false, a, false, b,
                                                 (short)0, c, false, false);
}

__device__ __forceinline__ float sigmoid_f(float x) {
  return 1.0f / (1.0f + __expf(-x));
}

// ---- fp32 -> bf16 weight conversion (once per launch) ----------------------
__global__ void __launch_bounds__(256)
convert_weights_bf16(const float* __restrict__ a, const float* __restrict__ b,
                     const float* __restrict__ c, __bf16* __restrict__ oa,
                     __bf16* __restrict__ ob, __bf16* __restrict__ oc, int n) {
  int i = blockIdx.x * blockDim.x + threadIdx.x;
  if (i < n) {
    oa[i] = (__bf16)a[i];
    ob[i] = (__bf16)b[i];
    oc[i] = (__bf16)c[i];
  }
}

// ---- layer-0 step ----------------------------------------------------------
// Block b: hidden columns n0 = b*16. Stages W_hh0 gate strips (48 KB) in LDS.
// Wave w: batch rows m0 = w*32 (two 16x16 sub-tiles).
__global__ void __launch_bounds__(256)
gru0_step(const float* __restrict__ x, int t,
          const float* __restrict__ Wih0,               // [3H, I] fp32
          const float* __restrict__ bih0, const float* __restrict__ bhh0,
          const __bf16* __restrict__ Whh0,              // [3H, H] bf16
          const float* __restrict__ hprev_f, const __bf16* __restrict__ hprev_b,
          float* __restrict__ hnew_f, __bf16* __restrict__ hnew_b) {
  const int H = GRU_H, T = GRU_T, I = GRU_I;
  extern __shared__ char smem_raw[];
  __bf16* sW = (__bf16*)smem_raw;                       // [3][16][512]

  int tid = threadIdx.x;
  int n0 = blockIdx.x * 16;

  // Stage 3 gate strips of W_hh0: 3*16*512 bf16 = 3072 16B chunks.
#pragma unroll
  for (int i = 0; i < 12; ++i) {
    int c = i * 256 + tid;
    int e = c * 8;                    // element offset in LDS image
    int g = e >> 13;                  // /8192: gate strip
    int rem = e & 8191;
    int sr = rem >> 9;                // /512: col within strip
    int k = rem & 511;
    cp16_to_lds(Whh0 + ((size_t)(g * H + n0 + sr)) * H + k, sW + e);
  }
  stage_fence();

  int wave = tid >> 5;
  int m0 = wave * 32;
  int lane = tid & 31;
  int ln = lane & 15, lh = lane >> 4;
  int col = n0 + ln;

  float bR  = bih0[col]       + bhh0[col];
  float bZ  = bih0[H + col]   + bhh0[H + col];
  float bNi = bih0[2 * H + col];
  float bNh = bhh0[2 * H + col];

  v8f accR[2], accZ[2], accN[2];
  float xwN[2][8];
#pragma unroll
  for (int rt = 0; rt < 2; ++rt) {
#pragma unroll
    for (int v = 0; v < 8; ++v) {
      int row = m0 + rt * 16 + v + 8 * lh;
      const float* xr = x + ((size_t)row * T + t) * I;
      float aR = 0.f, aZ = 0.f, aN = 0.f;
#pragma unroll
      for (int i = 0; i < I; ++i) {
        float xv = xr[i];
        aR += xv * Wih0[(size_t)col * I + i];
        aZ += xv * Wih0[((size_t)(H + col)) * I + i];
        aN += xv * Wih0[((size_t)(2 * H + col)) * I + i];
      }
      accR[rt][v] = aR + bR;
      accZ[rt][v] = aZ + bZ;
      accN[rt][v] = bNh;              // recurrent-n kept separate
      xwN[rt][v]  = aN + bNi;         // x-side n, added after r is known
    }
  }

#pragma unroll 2
  for (int kk = 0; kk < H; kk += 32) {
    v16bf a0 = load_a_frag(hprev_b, H, m0 + ln, kk, lh);
    v16bf a1 = load_a_frag(hprev_b, H, m0 + 16 + ln, kk, lh);
    v16bf br = load_b_lds(sW, 0, ln, kk, lh);
    v16bf bz = load_b_lds(sW, 1, ln, kk, lh);
    v16bf bn = load_b_lds(sW, 2, ln, kk, lh);
    accR[0] = wmma_bf16(a0, br, accR[0]);
    accR[1] = wmma_bf16(a1, br, accR[1]);
    accZ[0] = wmma_bf16(a0, bz, accZ[0]);
    accZ[1] = wmma_bf16(a1, bz, accZ[1]);
    accN[0] = wmma_bf16(a0, bn, accN[0]);
    accN[1] = wmma_bf16(a1, bn, accN[1]);
  }

#pragma unroll
  for (int rt = 0; rt < 2; ++rt) {
#pragma unroll
    for (int v = 0; v < 8; ++v) {
      int row = m0 + rt * 16 + v + 8 * lh;
      float r = sigmoid_f(accR[rt][v]);
      float z = sigmoid_f(accZ[rt][v]);
      float n = tanhf(xwN[rt][v] + r * accN[rt][v]);
      float ho = hprev_f[(size_t)row * H + col];
      float hn = (1.0f - z) * n + z * ho;
      hnew_f[(size_t)row * H + col] = hn;
      hnew_b[(size_t)row * H + col] = (__bf16)hn;
    }
  }
}

// ---- layer-1 step ----------------------------------------------------------
// Stages W_ih1 (strips 0..2) and W_hh1 (strips 3..5): 96 KB LDS.
__global__ void __launch_bounds__(256)
gru1_step(const __bf16* __restrict__ h1t_b,             // [B,H] layer-0 out @ t
          const __bf16* __restrict__ Wih1,              // [3H,H] bf16
          const __bf16* __restrict__ Whh1,              // [3H,H] bf16
          const float* __restrict__ bih1, const float* __restrict__ bhh1,
          const float* __restrict__ hprev_f, const __bf16* __restrict__ hprev_b,
          float* __restrict__ hnew_f, __bf16* __restrict__ hnew_b) {
  const int H = GRU_H;
  extern __shared__ char smem_raw[];
  __bf16* sW = (__bf16*)smem_raw;                       // [6][16][512]

  int tid = threadIdx.x;
  int n0 = blockIdx.x * 16;

  // 6 strips = 6144 16B chunks (24 per thread).
#pragma unroll
  for (int i = 0; i < 24; ++i) {
    int c = i * 256 + tid;
    int e = c * 8;
    int strip = e >> 13;              // 0..5
    int g = strip >= 3 ? strip - 3 : strip;
    const __bf16* src = (strip >= 3) ? Whh1 : Wih1;
    int rem = e & 8191;
    int sr = rem >> 9;
    int k = rem & 511;
    cp16_to_lds(src + ((size_t)(g * H + n0 + sr)) * H + k, sW + e);
  }
  stage_fence();

  int wave = tid >> 5;
  int m0 = wave * 32;
  int lane = tid & 31;
  int ln = lane & 15, lh = lane >> 4;
  int col = n0 + ln;

  float bR  = bih1[col] + bhh1[col];
  float bZ  = bih1[H + col] + bhh1[H + col];
  float bNi = bih1[2 * H + col];
  float bNh = bhh1[2 * H + col];

  v8f accR[2], accZ[2], accNx[2], accNh[2];
#pragma unroll
  for (int rt = 0; rt < 2; ++rt)
#pragma unroll
    for (int v = 0; v < 8; ++v) {
      accR[rt][v] = bR; accZ[rt][v] = bZ;
      accNx[rt][v] = bNi; accNh[rt][v] = bNh;
    }

  // Pass 1: input side (A = h1_t, B = W_ih1 strips 0..2)
#pragma unroll 2
  for (int kk = 0; kk < H; kk += 32) {
    v16bf a0 = load_a_frag(h1t_b, H, m0 + ln, kk, lh);
    v16bf a1 = load_a_frag(h1t_b, H, m0 + 16 + ln, kk, lh);
    v16bf br = load_b_lds(sW, 0, ln, kk, lh);
    v16bf bz = load_b_lds(sW, 1, ln, kk, lh);
    v16bf bn = load_b_lds(sW, 2, ln, kk, lh);
    accR[0]  = wmma_bf16(a0, br, accR[0]);
    accR[1]  = wmma_bf16(a1, br, accR[1]);
    accZ[0]  = wmma_bf16(a0, bz, accZ[0]);
    accZ[1]  = wmma_bf16(a1, bz, accZ[1]);
    accNx[0] = wmma_bf16(a0, bn, accNx[0]);
    accNx[1] = wmma_bf16(a1, bn, accNx[1]);
  }
  // Pass 2: recurrent side (A = h2_prev, B = W_hh1 strips 3..5)
#pragma unroll 2
  for (int kk = 0; kk < H; kk += 32) {
    v16bf a0 = load_a_frag(hprev_b, H, m0 + ln, kk, lh);
    v16bf a1 = load_a_frag(hprev_b, H, m0 + 16 + ln, kk, lh);
    v16bf br = load_b_lds(sW, 3, ln, kk, lh);
    v16bf bz = load_b_lds(sW, 4, ln, kk, lh);
    v16bf bn = load_b_lds(sW, 5, ln, kk, lh);
    accR[0]  = wmma_bf16(a0, br, accR[0]);
    accR[1]  = wmma_bf16(a1, br, accR[1]);
    accZ[0]  = wmma_bf16(a0, bz, accZ[0]);
    accZ[1]  = wmma_bf16(a1, bz, accZ[1]);
    accNh[0] = wmma_bf16(a0, bn, accNh[0]);
    accNh[1] = wmma_bf16(a1, bn, accNh[1]);
  }

#pragma unroll
  for (int rt = 0; rt < 2; ++rt)
#pragma unroll
    for (int v = 0; v < 8; ++v) {
      int row = m0 + rt * 16 + v + 8 * lh;
      float r = sigmoid_f(accR[rt][v]);
      float z = sigmoid_f(accZ[rt][v]);
      float n = tanhf(accNx[rt][v] + r * accNh[rt][v]);
      float ho = hprev_f[(size_t)row * H + col];
      float hn = (1.0f - z) * n + z * ho;
      hnew_f[(size_t)row * H + col] = hn;
      hnew_b[(size_t)row * H + col] = (__bf16)hn;
    }
}

// ---- head: out = h2_last @ W_fc^T + b_fc (6 MFLOP; plain VALU) -------------
__global__ void __launch_bounds__(256)
head_kernel(const float* __restrict__ h2, const float* __restrict__ Wfc,
            const float* __restrict__ bfc, float* __restrict__ out) {
  int i = blockIdx.x * blockDim.x + threadIdx.x;
  if (i >= GRU_B * GRU_OUT) return;
  int b = i / GRU_OUT, o = i % GRU_OUT;
  float acc = bfc[o];
#pragma unroll 8
  for (int k = 0; k < GRU_H; ++k)
    acc += h2[(size_t)b * GRU_H + k] * Wfc[(size_t)o * GRU_H + k];
  out[i] = acc;
}

// ---------------------------------------------------------------------------
extern "C" void kernel_launch(void* const* d_in, const int* in_sizes, int n_in,
                              void* d_out, int out_size, void* d_ws, size_t ws_size,
                              hipStream_t stream) {
  const float* x    = (const float*)d_in[0];
  const float* Wih0 = (const float*)d_in[1];
  const float* Whh0 = (const float*)d_in[2];
  const float* bih0 = (const float*)d_in[3];
  const float* bhh0 = (const float*)d_in[4];
  const float* Wih1 = (const float*)d_in[5];
  const float* Whh1 = (const float*)d_in[6];
  const float* bih1 = (const float*)d_in[7];
  const float* bhh1 = (const float*)d_in[8];
  const float* Wfc  = (const float*)d_in[9];
  const float* bfc  = (const float*)d_in[10];
  float* out = (float*)d_out;

  const int H = GRU_H, B = GRU_B, T = GRU_T;
  char* w = (char*)d_ws;
  auto carve = [&](size_t bytes) -> void* {
    void* p = (void*)w;
    w += (bytes + 255) & ~(size_t)255;
    return p;
  };

  const size_t wElems = (size_t)3 * H * H;   // 1536 x 512
  __bf16* Whh0b = (__bf16*)carve(wElems * 2);
  __bf16* Wih1b = (__bf16*)carve(wElems * 2);
  __bf16* Whh1b = (__bf16*)carve(wElems * 2);
  float*  h1f[2]; __bf16* h1b[2];
  float*  h2f[2]; __bf16* h2b[2];
  for (int i = 0; i < 2; ++i) {
    h1f[i] = (float*)carve((size_t)B * H * 4);
    h1b[i] = (__bf16*)carve((size_t)B * H * 2);
    h2f[i] = (float*)carve((size_t)B * H * 4);
    h2b[i] = (__bf16*)carve((size_t)B * H * 2);
  }

  {
    int n = (int)wElems;
    convert_weights_bf16<<<(n + 255) / 256, 256, 0, stream>>>(
        Whh0, Wih1, Whh1, Whh0b, Wih1b, Whh1b, n);
  }
  (void)hipMemsetAsync(h1f[0], 0, (size_t)B * H * 4, stream);
  (void)hipMemsetAsync(h1b[0], 0, (size_t)B * H * 2, stream);
  (void)hipMemsetAsync(h2f[0], 0, (size_t)B * H * 4, stream);
  (void)hipMemsetAsync(h2b[0], 0, (size_t)B * H * 2, stream);

  const size_t lds0 = (size_t)3 * 16 * H * 2;   // 48 KB
  const size_t lds1 = (size_t)6 * 16 * H * 2;   // 96 KB
  int cur = 0;
  for (int t = 0; t < T; ++t) {
    int nxt = cur ^ 1;
    gru0_step<<<32, 256, lds0, stream>>>(x, t, Wih0, bih0, bhh0, Whh0b,
                                         h1f[cur], h1b[cur], h1f[nxt], h1b[nxt]);
    gru1_step<<<32, 256, lds1, stream>>>(h1b[nxt], Wih1b, Whh1b, bih1, bhh1,
                                         h2f[cur], h2b[cur], h2f[nxt], h2b[nxt]);
    cur = nxt;
  }
  head_kernel<<<(GRU_B * GRU_OUT + 255) / 256, 256, 0, stream>>>(
      h2f[cur], Wfc, bfc, out);
}
